// MultiHeadSelfAttention2D_23691039605088
// MI455X (gfx1250) — matrix operand order repaired and checked by reading, not compiled, not run on hardware
//
#include <hip/hip_runtime.h>
#include <hip/hip_bf16.h>

// ---------------------------------------------------------------------------
// MHSA 2D: B=64, C=512, H=W=16 (N=256), heads=8, hd=64
// cvt/transpose (LDS-tiled) ; QKV GEMM (WMMA f16->f32, LDS-staged coalesced
// epilogue) ; fused bias+softmax+PV attention ; out-proj GEMM.
// All matmuls via v_wmma_f32_16x16x32_f16 (fp32 accumulate).
// ---------------------------------------------------------------------------

#define B_ 64
#define C_ 512
#define N_ 256
#define HEADS_ 8
#define HD_ 64
#define M1_ 1536   // 3*C

typedef __attribute__((ext_vector_type(16))) _Float16 v16h;
typedef __attribute__((ext_vector_type(8)))  _Float16 v8h;
typedef __attribute__((ext_vector_type(4)))  _Float16 v4h;
typedef __attribute__((ext_vector_type(8)))  float    v8f;

// A-matrix 16x32 f16 fragment (ISA 7.12.2): lane l -> row row0+(l&15),
// halves at k0+8g and k0+16+8g (g = l>>4). Two 16B loads per lane.
static __device__ __forceinline__ v16h load_a16x32(const _Float16* base, int row0,
                                                   int lda, int k0, int lane) {
  int g = lane >> 4;
  const _Float16* p = base + (size_t)(row0 + (lane & 15)) * lda + k0 + 8 * g;
  union { v16h v; v8h h[2]; } f;
  f.h[0] = *(const v8h*)(p);
  f.h[1] = *(const v8h*)(p + 16);
  return f.v;
}

// B-matrix 32x16 f16 fragment from an N-major (transposed) buffer:
// lane l -> column col0+(l&15); halves at k0+8g and k0+16+8g (g = l>>4).
static __device__ __forceinline__ v16h load_b32x16(const _Float16* baseT, int col0,
                                                   int ldb, int k0, int lane) {
  int g = lane >> 4;
  const _Float16* p = baseT + (size_t)(col0 + (lane & 15)) * ldb + k0 + 8 * g;
  union { v16h v; v8h h[2]; } f;
  f.h[0] = *(const v8h*)(p);
  f.h[1] = *(const v8h*)(p + 16);
  return f.v;
}

static __device__ __forceinline__ v8f wmma_f16(v16h a, v16h b, v8f c) {
  return __builtin_amdgcn_wmma_f32_16x16x32_f16(false, a, false, b, (short)0, c,
                                                false, false);
}

static __device__ __forceinline__ v8f zero8() {
  v8f z = {0.f, 0.f, 0.f, 0.f, 0.f, 0.f, 0.f, 0.f};
  return z;
}

// ---------------------------------------------------------------------------
// Stage 0a: weight conversion (contiguous, already coalesced)
// ---------------------------------------------------------------------------
__global__ void cvt_f32_to_f16(const float* __restrict__ src,
                               _Float16* __restrict__ dst, int n) {
  int i = blockIdx.x * 256 + threadIdx.x;
  if (i < n) dst[i] = (_Float16)src[i];
}

// ---------------------------------------------------------------------------
// Stage 0b: x [B][C][N] f32 -> Xt [B][N][C] f16, 32x32 LDS tile transpose.
// grid = B * (C/32) * (N/32) = 64*16*8 ; reads float4-coalesced,
// writes v4h (8B) chunks -> 64B-contiguous rows.
// ---------------------------------------------------------------------------
__global__ void transpose_x_f16(const float* __restrict__ x,
                                _Float16* __restrict__ Xt) {
  __shared__ _Float16 tx[32][33];
  int bx = blockIdx.x;
  int b  = bx >> 7;
  int r  = bx & 127;
  int c0 = (r >> 3) * 32;          // 16 c-tiles
  int n0 = (r & 7) * 32;           // 8 n-tiles
  int t  = threadIdx.x;

  // read: thread -> (ci, 4 consecutive n)
  int ci  = t >> 3;
  int nj4 = (t & 7) * 4;
  const float4 f4 = *(const float4*)(x + ((size_t)(b * C_ + c0 + ci) * N_) + n0 + nj4);
  tx[ci][nj4 + 0] = (_Float16)f4.x;
  tx[ci][nj4 + 1] = (_Float16)f4.y;
  tx[ci][nj4 + 2] = (_Float16)f4.z;
  tx[ci][nj4 + 3] = (_Float16)f4.w;
  __syncthreads();

  // write: thread -> (n row, 4 consecutive c)
  int nr  = t >> 3;
  int cj4 = (t & 7) * 4;
  v4h o;
  o[0] = tx[cj4 + 0][nr];
  o[1] = tx[cj4 + 1][nr];
  o[2] = tx[cj4 + 2][nr];
  o[3] = tx[cj4 + 3][nr];
  *(v4h*)(Xt + ((size_t)(b * N_ + n0 + nr) * C_) + c0 + cj4) = o;
}

// ---------------------------------------------------------------------------
// Stage 1: QKV projection.  per batch: (1536x512) @ (512x256)
// block = 256 thr (8 waves), block tile 128M x 64N, wave tile 32x32.
// grid.x = 64 * 12 * 4.  Epilogue: +qkv_b -> LDS tile -> coalesced v8h
// stores into Qt[b][h][n][d], Kt[b][h][n][d], V[b][h][d][n].
// A 128-row M-tile = one of {q,k,v} x 2 full heads, so remap is uniform.
// ---------------------------------------------------------------------------
__global__ void qkv_gemm(const _Float16* __restrict__ Wq,   // [1536][512]
                         const _Float16* __restrict__ Xt,   // [B][256][512]
                         const float* __restrict__ qkv_b,   // [1536]
                         _Float16* __restrict__ Qt,         // [B][8][256][64]
                         _Float16* __restrict__ Kt,         // [B][8][256][64]
                         _Float16* __restrict__ Vb)         // [B][8][64][256]
{
  __shared__ _Float16 tileS[128][72];   // [o_local][n_local], 16B-mult pitch
  int bx = blockIdx.x;
  int b  = bx / (12 * 4);
  int r  = bx % (12 * 4);
  int mt = r / 4, nt = r % 4;
  int lane = threadIdx.x & 31;
  int w    = threadIdx.x >> 5;
  int Mb = mt * 128, Nb = nt * 64;
  int M0 = Mb + (w >> 1) * 32;
  int N0 = Nb + (w & 1) * 32;
  const _Float16* Xb = Xt + (size_t)b * N_ * C_;

  v8f acc[2][2];
  acc[0][0] = zero8(); acc[0][1] = zero8();
  acc[1][0] = zero8(); acc[1][1] = zero8();

  for (int k0 = 0; k0 < C_; k0 += 32) {
    v16h a0 = load_a16x32(Wq, M0,      C_, k0, lane);
    v16h a1 = load_a16x32(Wq, M0 + 16, C_, k0, lane);
    v16h b0 = load_b32x16(Xb, N0,      C_, k0, lane);
    v16h b1 = load_b32x16(Xb, N0 + 16, C_, k0, lane);
    acc[0][0] = wmma_f16(a0, b0, acc[0][0]);
    acc[0][1] = wmma_f16(a0, b1, acc[0][1]);
    acc[1][0] = wmma_f16(a1, b0, acc[1][0]);
    acc[1][1] = wmma_f16(a1, b1, acc[1][1]);
  }

  // D-frag -> LDS (with qkv bias)
  int g = lane >> 4;
  for (int i = 0; i < 2; ++i)
    for (int j = 0; j < 2; ++j)
      for (int rr = 0; rr < 8; ++rr) {
        int ol = (w >> 1) * 32 + i * 16 + rr + 8 * g;   // 0..127
        int nl = (w & 1) * 32 + j * 16 + (lane & 15);   // 0..63
        tileS[ol][nl] = (_Float16)(acc[i][j][rr] + qkv_b[Mb + ol]);
      }
  __syncthreads();

  // coalesced store phase: 256 threads -> 128 rows x 2 halves of 32 f16
  int which = mt >> 2;                // 0=q 1=k 2=v
  int hpair = mt & 3;
  int t    = threadIdx.x;
  int ri   = t >> 1;                  // 0..127
  int half = t & 1;
  int hl   = ri >> 6;
  int idx  = ri & 63;
  size_t bh = (size_t)b * HEADS_ + (hpair * 2 + hl);

  if (which < 2) {
    // dst row = token n, contiguous d (128B rows)
    _Float16* dst = (which == 0 ? Qt : Kt) + (bh * N_ + Nb + idx) * HD_ + half * 32;
    _Float16 tmp[32];
    for (int d = 0; d < 32; ++d) tmp[d] = tileS[hl * 64 + half * 32 + d][idx];
    for (int q = 0; q < 4; ++q) *(v8h*)(dst + q * 8) = *(const v8h*)(tmp + q * 8);
  } else {
    // dst row = d, contiguous n (LDS rows are n-contiguous too)
    _Float16* dst = Vb + (bh * HD_ + idx) * N_ + Nb + half * 32;
    for (int q = 0; q < 4; ++q)
      *(v8h*)(dst + q * 8) = *(const v8h*)(&tileS[hl * 64 + idx][half * 32 + q * 8]);
  }
}

// ---------------------------------------------------------------------------
// Stage 2: fused attention.  One block per (b, head, 64-row chunk).
// S = scale*Qt.Kt^T + bias -> LDS (f32) ; softmax ; O^T = P @ V^T -> AOt f16
// dynamic LDS: 64*256*4 (S) + 64*256*2 (P) + 4096 (rel) = 102400 B
// ---------------------------------------------------------------------------
__global__ void attention_fused(const _Float16* __restrict__ Qt,
                                const _Float16* __restrict__ Kt,
                                const _Float16* __restrict__ Vb,
                                const float* __restrict__ rel,   // [8][31][31]
                                _Float16* __restrict__ AOt)      // [B][256][512]
{
  extern __shared__ char smem[];
  float*    S  = (float*)smem;                            // [64][256]
  _Float16* P  = (_Float16*)(smem + 64 * 256 * 4);        // [64][256]
  float*    RB = (float*)(smem + 64 * 256 * 4 + 64 * 256 * 2); // [961]

  int bx    = blockIdx.x;          // B*HEADS*4
  int b     = bx >> 5;
  int rem   = bx & 31;
  int head  = rem >> 2;
  int chunk = rem & 3;
  int n0    = chunk * 64;
  int tid   = threadIdx.x;
  int lane  = tid & 31;
  int w     = tid >> 5;
  int g     = lane >> 4;

  for (int i = tid; i < 31 * 31; i += 256) RB[i] = rel[head * 31 * 31 + i];
  __syncthreads();

  size_t bh = (size_t)b * HEADS_ + head;
  const _Float16* Qbh = Qt + bh * N_ * HD_;
  const _Float16* Kbh = Kt + bh * N_ * HD_;
  const _Float16* Vbh = Vb + bh * HD_ * N_;
  const float scale = 0.125f;   // hd^-0.5

  // ---- phase 1: S tiles. wave w -> row-tile rt=w>>1, 8 col-tiles
  int rt = w >> 1;
  for (int cc = 0; cc < 8; ++cc) {
    int ct = (w & 1) * 8 + cc;
    v8f acc = zero8();
    for (int k0 = 0; k0 < HD_; k0 += 32) {
      v16h a  = load_a16x32(Qbh, n0 + rt * 16, HD_, k0, lane);
      v16h bb = load_b32x16(Kbh, ct * 16, HD_, k0, lane);
      acc = wmma_f16(a, bb, acc);
    }
    int m  = ct * 16 + (lane & 15);
    int h2 = m >> 4, w2 = m & 15;
    for (int rr = 0; rr < 8; ++rr) {
      int nl = rt * 16 + rr + 8 * g;
      int ng = n0 + nl;
      int h1 = ng >> 4, w1 = ng & 15;
      float bias = RB[(h1 - h2 + 15) * 31 + (w1 - w2 + 15)];
      S[nl * 256 + m] = acc[rr] * scale + bias;
    }
  }
  __syncthreads();

  // ---- phase 2: row softmax, 8 rows per wave
  for (int rr = 0; rr < 8; ++rr) {
    int row = w * 8 + rr;
    float vals[8];
    float mx = -3.0e38f;
    for (int j = 0; j < 8; ++j) {
      vals[j] = S[row * 256 + lane + 32 * j];
      mx = fmaxf(mx, vals[j]);
    }
    for (int off = 16; off >= 1; off >>= 1) mx = fmaxf(mx, __shfl_xor(mx, off, 32));
    float sum = 0.f;
    for (int j = 0; j < 8; ++j) { vals[j] = __expf(vals[j] - mx); sum += vals[j]; }
    for (int off = 16; off >= 1; off >>= 1) sum += __shfl_xor(sum, off, 32);
    float inv = 1.f / sum;
    for (int j = 0; j < 8; ++j)
      P[row * 256 + lane + 32 * j] = (_Float16)(vals[j] * inv);
  }
  __syncthreads();

  // ---- phase 3: O^T(64 x 64) = P(64x256) @ V^T ; A-frags from LDS (ds_load)
  for (int q = 0; q < 2; ++q) {
    int ct2 = (w & 1) * 2 + q;     // d-tile 0..3
    v8f acc = zero8();
    for (int k0 = 0; k0 < N_; k0 += 32) {
      v16h a  = load_a16x32((const _Float16*)P, rt * 16, 256, k0, lane);
      v16h bb = load_b32x16(Vbh, ct2 * 16, N_, k0, lane);
      acc = wmma_f16(a, bb, acc);
    }
    int d = ct2 * 16 + (lane & 15);
    for (int rr = 0; rr < 8; ++rr) {
      int ng = n0 + rt * 16 + rr + 8 * g;
      // lanes 0..15 write 16 consecutive d -> 32B-contiguous
      AOt[((size_t)b * N_ + ng) * C_ + head * HD_ + d] = (_Float16)acc[rr];
    }
  }
}

// ---------------------------------------------------------------------------
// Stage 3: output projection. per batch (512x512) @ (512x256), +out_b, f32 out
// grid.x = 64 * 4 * 4 ; f32 stores are lane-contiguous (64B/half-wave).
// ---------------------------------------------------------------------------
__global__ void out_gemm(const _Float16* __restrict__ Wo,   // [512][512]
                         const _Float16* __restrict__ AOt,  // [B][256][512]
                         const float* __restrict__ out_b,   // [512]
                         float* __restrict__ y)             // [B][512][256]
{
  int bx = blockIdx.x;
  int b  = bx / 16;
  int r  = bx % 16;
  int mt = r / 4, nt = r % 4;
  int lane = threadIdx.x & 31;
  int w    = threadIdx.x >> 5;
  int M0 = mt * 128 + (w >> 1) * 32;
  int N0 = nt * 64  + (w & 1) * 32;
  const _Float16* Ab = AOt + (size_t)b * N_ * C_;

  v8f acc[2][2];
  acc[0][0] = zero8(); acc[0][1] = zero8();
  acc[1][0] = zero8(); acc[1][1] = zero8();

  for (int k0 = 0; k0 < C_; k0 += 32) {
    v16h a0 = load_a16x32(Wo, M0,      C_, k0, lane);
    v16h a1 = load_a16x32(Wo, M0 + 16, C_, k0, lane);
    v16h b0 = load_b32x16(Ab, N0,      C_, k0, lane);
    v16h b1 = load_b32x16(Ab, N0 + 16, C_, k0, lane);
    acc[0][0] = wmma_f16(a0, b0, acc[0][0]);
    acc[0][1] = wmma_f16(a0, b1, acc[0][1]);
    acc[1][0] = wmma_f16(a1, b0, acc[1][0]);
    acc[1][1] = wmma_f16(a1, b1, acc[1][1]);
  }

  int g = lane >> 4;
  for (int i = 0; i < 2; ++i)
    for (int j = 0; j < 2; ++j)
      for (int rr = 0; rr < 8; ++rr) {
        int o = M0 + i * 16 + rr + 8 * g;
        int n = N0 + j * 16 + (lane & 15);
        y[((size_t)b * C_ + o) * N_ + n] = acc[i][j][rr] + out_b[o];
      }
}

// ---------------------------------------------------------------------------
// launch
// ---------------------------------------------------------------------------
extern "C" void kernel_launch(void* const* d_in, const int* in_sizes, int n_in,
                              void* d_out, int out_size, void* d_ws, size_t ws_size,
                              hipStream_t stream) {
  const float* x      = (const float*)d_in[0];
  const float* qkv_w  = (const float*)d_in[1];
  const float* qkv_b  = (const float*)d_in[2];
  const float* out_w  = (const float*)d_in[3];
  const float* out_b  = (const float*)d_in[4];
  const float* rel    = (const float*)d_in[5];
  float* y = (float*)d_out;

  // workspace layout (f16 scratch, ~86 MB total)
  char* ws = (char*)d_ws;
  _Float16* Xt  = (_Float16*)(ws);                         // 64*256*512
  _Float16* Wq  = (_Float16*)(ws + 16777216);              // 1536*512
  _Float16* Wo  = (_Float16*)(ws + 18350080);              // 512*512
  _Float16* Qt  = (_Float16*)(ws + 18874368);              // 64*8*256*64
  _Float16* Kt  = (_Float16*)(ws + 35651584);
  _Float16* Vb  = (_Float16*)(ws + 52428800);
  _Float16* AOt = (_Float16*)(ws + 69206016);              // 64*256*512

  // stage 0: conversions
  cvt_f32_to_f16<<<(M1_ * C_) / 256, 256, 0, stream>>>(qkv_w, Wq, M1_ * C_);
  cvt_f32_to_f16<<<(C_ * C_) / 256, 256, 0, stream>>>(out_w, Wo, C_ * C_);
  transpose_x_f16<<<B_ * 16 * 8, 256, 0, stream>>>(x, Xt);

  // stage 1: QKV projection
  qkv_gemm<<<B_ * 12 * 4, 256, 0, stream>>>(Wq, Xt, qkv_b, Qt, Kt, Vb);

  // stage 2: fused attention (dynamic LDS 100 KB)
  attention_fused<<<B_ * HEADS_ * 4, 256, 102400, stream>>>(Qt, Kt, Vb, rel, AOt);

  // stage 3: output projection
  out_gemm<<<B_ * 4 * 4, 256, 0, stream>>>(Wo, AOt, out_b, y);
}
